// Fusion_T_26998164423019
// MI455X (gfx1250) — matrix operand order repaired and checked by reading, not compiled, not run on hardware
//
#include <hip/hip_runtime.h>
#include <hip/hip_bf16.h>
#include <math.h>
#include <stdint.h>

// ---------------------------------------------------------------------------
// Types for CDNA5 WMMA (wave32)
// ---------------------------------------------------------------------------
typedef __bf16 bf16_t;
typedef __attribute__((ext_vector_type(16))) bf16_t v16bf;
typedef __attribute__((ext_vector_type(8)))  bf16_t v8bf;
typedef __attribute__((ext_vector_type(8)))  float  v8f;

#define DIM        576
#define SEQ        1024
#define BATCH      8
#define ROWS       (BATCH * SEQ)          /* 8192 */
#define NBLK       6
#define CAT_DIM    (NBLK * DIM)           /* 3456 */
#define LN_EPS     1e-5f
#define ATTN_SCALE 0.041666666666666664f  /* 576^-0.5 */

// ---------------------------------------------------------------------------
// WMMA fragment loaders (bf16, 16x16x32)
//
// A matrix 16x32 (MxK), row-major source [rows x K]:
//   lane<16:  row=lane,    K = {0..7} u {16..23}  (elements 0..7, 8..15)
//   lane>=16: row=lane-16, K = {8..15} u {24..31}
// B matrix 32x16 (KxN) supplied as W[N x K] row-major (i.e. B^T):
//   N = lane & 15, same K split as A (lane<16: {0..7} u {16..23}, else {8..15} u {24..31})
// ---------------------------------------------------------------------------
__device__ __forceinline__ v16bf load_frag_a(const bf16_t* __restrict__ p0,
                                             int ld, int lane) {
  const int r  = lane & 15;
  const int kb = (lane >> 4) << 3;             // 0 or 8
  const bf16_t* p = p0 + (size_t)r * ld + kb;
  v8bf lo = *(const v8bf*)(p);                 // K = kb .. kb+7
  v8bf hi = *(const v8bf*)(p + 16);            // K = kb+16 .. kb+23
  v16bf f;
#pragma unroll
  for (int i = 0; i < 8; ++i) { f[i] = lo[i]; f[i + 8] = hi[i]; }
  return f;
}

__device__ __forceinline__ v16bf load_frag_b(const bf16_t* p0, int ld,
                                             int lane) {
  const int n  = lane & 15;
  const int kb = (lane >> 4) << 3;             // 0 or 8
  const bf16_t* p = p0 + (size_t)n * ld + kb;
  v8bf lo = *(const v8bf*)(p);                 // K = kb .. kb+7
  v8bf hi = *(const v8bf*)(p + 16);            // K = kb+16 .. kb+23
  v16bf f;
#pragma unroll
  for (int i = 0; i < 8; ++i) { f[i] = lo[i]; f[i + 8] = hi[i]; }
  return f;
}

__device__ __forceinline__ v8f wmma_bf16(v16bf a, v16bf b, v8f c) {
  return __builtin_amdgcn_wmma_f32_16x16x32_bf16(
      /*neg_a=*/false, a, /*neg_b=*/false, b,
      /*c_mod=*/(short)0, c, /*reuse_a=*/false, /*reuse_b=*/false);
}

// ---------------------------------------------------------------------------
// gfx1250 async global->LDS copy (ASYNCcnt), 16 bytes per lane.
// Inline asm: portable across toolchains (builtin arity differs).
// ---------------------------------------------------------------------------
__device__ __forceinline__ void async_copy_b128(uint32_t lds_byte_off,
                                                const bf16_t* gaddr) {
  asm volatile("global_load_async_to_lds_b128 %0, %1, off"
               :
               : "v"(lds_byte_off), "v"(gaddr)
               : "memory");
}
__device__ __forceinline__ void wait_async0() {
  asm volatile("s_wait_asynccnt 0x0" ::: "memory");
}

// ---------------------------------------------------------------------------
// Generic GEMM: out = A[M x K](bf16) @ W[O x K]^T(bf16)  (+ epilogue)
//
// Workgroup = 8 waves = 256x64 output tile; each wave owns 32 rows.
// The 64x32 weight panel (4 KB) is staged in LDS via async copies
// (one b128 per thread per K-step), double-buffered so stage s+1's copy
// overlaps stage s's 8 WMMAs; all 8 waves share the panel via ds_load_b128.
//
// MODE 0: store bf16 row-major                (Q, K projections, no bias)
// MODE 1: store bf16 transposed per batch     (V -> Vt[b][576][1024])
// MODE 2: store f32 + bias + LN(x) residual   (P projection)
// MODE 3: store f32 + bias + exact GELU       (final fusion)
// ---------------------------------------------------------------------------
template <int MODE>
__global__ __launch_bounds__(256) void gemm_wmma_kernel(
    const bf16_t* __restrict__ A, int lda,
    const bf16_t* __restrict__ W, int ldw,
    const float* __restrict__ bias,
    float* __restrict__ outF, bf16_t* __restrict__ outB,
    const float* __restrict__ xsrc, const float* __restrict__ stats,
    const float* __restrict__ lng, const float* __restrict__ lnb,
    int K, int tiles_n) {
  __shared__ bf16_t Bsm[2][64 * 32];           // double-buffered weight panel
  const int tid  = threadIdx.x;
  const int lane = tid & 31;
  const int wave = tid >> 5;
  const int tm   = (int)blockIdx.x / tiles_n;
  const int tn   = (int)blockIdx.x % tiles_n;
  const int m0w  = tm * 256 + wave * 32;       // this wave's 32 rows
  const int n0   = tn * 64;
  const int ncols = tiles_n * 64;

  // async staging: thread -> (panel row, 16B chunk)
  const int srow   = tid >> 2;                 // 0..63
  const int schunk = tid & 3;                  // 0..3
  const bf16_t* wbase = W + (size_t)(n0 + srow) * ldw + schunk * 8;
  uint32_t lds_off[2];
  lds_off[0] = (uint32_t)(uintptr_t)&Bsm[0][0] +
               (uint32_t)(srow * 64 + schunk * 16);
  lds_off[1] = (uint32_t)(uintptr_t)&Bsm[1][0] +
               (uint32_t)(srow * 64 + schunk * 16);

  v8f zero = {};
  v8f acc[2][4];
#pragma unroll
  for (int i = 0; i < 2; ++i)
#pragma unroll
    for (int j = 0; j < 4; ++j) acc[i][j] = zero;

  const int nst = K / 32;

  // prologue: stage 0
  async_copy_b128(lds_off[0], wbase);
  wait_async0();
  __syncthreads();

  for (int s = 0; s < nst; ++s) {
    const int cur = s & 1;
    const int k0  = s * 32;
    if (s + 1 < nst)                           // overlap next panel copy
      async_copy_b128(lds_off[cur ^ 1], wbase + (size_t)(k0 + 32));

    // compute from current panel
    const bf16_t* bufp = &Bsm[cur][0];
    v16bf a0 = load_frag_a(A + (size_t)m0w * lda + k0, lda, lane);
    v16bf a1 = load_frag_a(A + (size_t)(m0w + 16) * lda + k0, lda, lane);
    v16bf b0 = load_frag_b(bufp + 0 * 16 * 32, 32, lane);
    v16bf b1 = load_frag_b(bufp + 1 * 16 * 32, 32, lane);
    v16bf b2 = load_frag_b(bufp + 2 * 16 * 32, 32, lane);
    v16bf b3 = load_frag_b(bufp + 3 * 16 * 32, 32, lane);
    acc[0][0] = wmma_bf16(a0, b0, acc[0][0]);
    acc[0][1] = wmma_bf16(a0, b1, acc[0][1]);
    acc[0][2] = wmma_bf16(a0, b2, acc[0][2]);
    acc[0][3] = wmma_bf16(a0, b3, acc[0][3]);
    acc[1][0] = wmma_bf16(a1, b0, acc[1][0]);
    acc[1][1] = wmma_bf16(a1, b1, acc[1][1]);
    acc[1][2] = wmma_bf16(a1, b2, acc[1][2]);
    acc[1][3] = wmma_bf16(a1, b3, acc[1][3]);

    if (s + 1 < nst) wait_async0();            // next panel resident
    __syncthreads();                           // all waves done with cur panel
  }

  // D layout: VGPR v -> M = v (lanes 0-15) / v+8 (lanes 16-31); N = lane & 15
  const int mloc = (lane >> 4) << 3;
  const int nc   = lane & 15;
#pragma unroll
  for (int i = 0; i < 2; ++i)
#pragma unroll
    for (int j = 0; j < 4; ++j)
#pragma unroll
      for (int v = 0; v < 8; ++v) {
        const int row = m0w + i * 16 + mloc + v;
        const int col = n0 + j * 16 + nc;
        const float val = acc[i][j][v];
        if (MODE == 0) {
          outB[(size_t)row * ncols + col] = (bf16_t)val;
        } else if (MODE == 1) {
          const int b = row >> 10;             // batch
          const int n = row & (SEQ - 1);       // seq pos
          outB[(size_t)b * (DIM * SEQ) + (size_t)col * SEQ + n] = (bf16_t)val;
        } else if (MODE == 2) {
          const float mu = stats[2 * row];
          const float rs = stats[2 * row + 1];
          const float xn =
              (xsrc[(size_t)row * DIM + col] - mu) * rs * lng[col] + lnb[col];
          outF[(size_t)row * ncols + col] = val + bias[col] + xn;
        } else {
          const float y = val + bias[col];
          outF[(size_t)row * ncols + col] =
              0.5f * y * (1.0f + erff(y * 0.70710678118654752f));
        }
      }
}

// ---------------------------------------------------------------------------
// Sigmoid cross-attention: one workgroup = one 16-row Q tile of one batch.
// Phase 1: S = sigmoid(Q K^T * scale) -> bf16 in LDS [16 x 1024]  (32 KB)
// Phase 2: O = S @ V with A-fragments from LDS, B-fragments from Vt
// ---------------------------------------------------------------------------
__global__ __launch_bounds__(256) void attn_kernel(
    const bf16_t* __restrict__ Q, const bf16_t* __restrict__ Km,
    const bf16_t* __restrict__ Vt, bf16_t* __restrict__ O) {
  __shared__ bf16_t S[16 * SEQ];
  const int lane = threadIdx.x & 31;
  const int wave = threadIdx.x >> 5;
  const int qt   = blockIdx.x;   // 0..63
  const int b    = blockIdx.y;   // 0..7
  const int mloc = (lane >> 4) << 3;
  const int nc   = lane & 15;

  const bf16_t* qbase = Q + (size_t)(b * SEQ + qt * 16) * DIM;
  v8f zero = {};

  // ---- phase 1: scores ----
  for (int nt = wave; nt < 64; nt += 8) {
    v8f acc = zero;
    const bf16_t* kbase = Km + (size_t)(b * SEQ + nt * 16) * DIM;
    for (int k0 = 0; k0 < DIM; k0 += 32) {
      v16bf af = load_frag_a(qbase + k0, DIM, lane);
      v16bf bf_ = load_frag_b(kbase + k0, DIM, lane);
      acc = wmma_bf16(af, bf_, acc);
    }
#pragma unroll
    for (int v = 0; v < 8; ++v) {
      float s = acc[v] * ATTN_SCALE;
      s = 1.0f / (1.0f + __expf(-s));
      S[(mloc + v) * SEQ + nt * 16 + nc] = (bf16_t)s;
    }
  }
  __syncthreads();

  // ---- phase 2: O = S @ V ----
  const int r   = lane & 15;
  const int kba = (lane >> 4) << 3;            // A-fragment K base (0/8)
  for (int ct = wave; ct < 36; ct += 8) {      // 36 channel tiles of 16
    v8f acc = zero;
    const bf16_t* vtb = Vt + (size_t)b * (DIM * SEQ) + (size_t)(ct * 16) * SEQ;
    for (int kp = 0; kp < 32; ++kp) {          // 32 key chunks of 32
      const bf16_t* p = &S[r * SEQ + kp * 32 + kba];
      v8bf lo = *(const v8bf*)(p);
      v8bf hi = *(const v8bf*)(p + 16);
      v16bf af;
#pragma unroll
      for (int i = 0; i < 8; ++i) { af[i] = lo[i]; af[i + 8] = hi[i]; }
      v16bf bf_ = load_frag_b(vtb + kp * 32, SEQ, lane);
      acc = wmma_bf16(af, bf_, acc);
    }
#pragma unroll
    for (int v = 0; v < 8; ++v) {
      const int row = b * SEQ + qt * 16 + mloc + v;
      const int col = ct * 16 + nc;
      O[(size_t)row * DIM + col] = (bf16_t)acc[v];
    }
  }
}

// ---------------------------------------------------------------------------
// LayerNorm over C=576: one wave per row (18 elems/lane), wave32 xor-reduce.
// Writes bf16 (g,b applied); optionally writes (mu, rstd) stats.
// ---------------------------------------------------------------------------
__global__ __launch_bounds__(256) void ln_kernel(
    const float* __restrict__ src, int ld_src,
    const float* __restrict__ g, const float* __restrict__ bb,
    bf16_t* __restrict__ dst, int ld_dst,
    float* __restrict__ stats, int rows) {
  const int lane = threadIdx.x & 31;
  const int row  = blockIdx.x * 8 + (threadIdx.x >> 5);
  if (row >= rows) return;
  const float* p = src + (size_t)row * ld_src;
  float v[18];
  float s = 0.f;
#pragma unroll
  for (int i = 0; i < 18; ++i) { v[i] = p[i * 32 + lane]; s += v[i]; }
#pragma unroll
  for (int off = 16; off; off >>= 1) s += __shfl_xor(s, off, 32);
  const float mu = s * (1.0f / 576.0f);
  float q = 0.f;
#pragma unroll
  for (int i = 0; i < 18; ++i) { const float d = v[i] - mu; q += d * d; }
#pragma unroll
  for (int off = 16; off; off >>= 1) q += __shfl_xor(q, off, 32);
  const float rstd = rsqrtf(q * (1.0f / 576.0f) + LN_EPS);
  if (stats != nullptr && lane == 0) {
    stats[2 * row] = mu;
    stats[2 * row + 1] = rstd;
  }
  bf16_t* d = dst + (size_t)row * ld_dst;
#pragma unroll
  for (int i = 0; i < 18; ++i) {
    const int c = i * 32 + lane;
    d[c] = (bf16_t)((v[i] - mu) * rstd * g[c] + bb[c]);
  }
}

// ---------------------------------------------------------------------------
// f32 -> bf16 grid-stride conversion (weights)
// ---------------------------------------------------------------------------
__global__ void cvt_bf16_kernel(const float* __restrict__ s,
                                bf16_t* __restrict__ d, int n) {
  int i = blockIdx.x * blockDim.x + threadIdx.x;
  const int stride = gridDim.x * blockDim.x;
  for (; i < n; i += stride) d[i] = (bf16_t)s[i];
}

// ---------------------------------------------------------------------------
// Host orchestration
// ---------------------------------------------------------------------------
extern "C" void kernel_launch(void* const* d_in, const int* in_sizes, int n_in,
                              void* d_out, int out_size, void* d_ws,
                              size_t ws_size, hipStream_t stream) {
  (void)in_sizes; (void)n_in; (void)out_size; (void)ws_size;

  const float* image = (const float*)d_in[0];
  const float* aolp  = (const float*)d_in[1];
  const float* dolp  = (const float*)d_in[2];
  const float* lnx_g = (const float*)d_in[3];
  const float* lnx_b = (const float*)d_in[4];
  const float* lny_g = (const float*)d_in[5];
  const float* lny_b = (const float*)d_in[6];
  const float* lnz_g = (const float*)d_in[7];
  const float* lnz_b = (const float*)d_in[8];
  const float* Wq    = (const float*)d_in[9];
  const float* Wk    = (const float*)d_in[10];
  const float* Wv    = (const float*)d_in[11];
  const float* Wp    = (const float*)d_in[12];
  const float* bp    = (const float*)d_in[13];
  const float* Wf    = (const float*)d_in[14];
  const float* bfv   = (const float*)d_in[15];

  char* ws = (char*)d_ws;
  size_t off = 0;
  auto wsalloc = [&](size_t bytes) -> char* {
    char* p = ws + off;
    off = (off + bytes + 255) & ~(size_t)255;
    return p;
  };

  const size_t WMAT = (size_t)NBLK * DIM * DIM;         // 1,990,656 elems
  bf16_t* wq_b = (bf16_t*)wsalloc(WMAT * 2);
  bf16_t* wk_b = (bf16_t*)wsalloc(WMAT * 2);
  bf16_t* wv_b = (bf16_t*)wsalloc(WMAT * 2);
  bf16_t* wp_b = (bf16_t*)wsalloc(WMAT * 2);
  bf16_t* wf_b = (bf16_t*)wsalloc((size_t)DIM * CAT_DIM * 2);
  bf16_t* XN   = (bf16_t*)wsalloc((size_t)ROWS * DIM * 2);
  bf16_t* YN   = (bf16_t*)wsalloc((size_t)ROWS * DIM * 2);
  float*  STATS= (float*)wsalloc((size_t)ROWS * 2 * 4);
  bf16_t* QB   = (bf16_t*)wsalloc((size_t)ROWS * DIM * 2);
  bf16_t* KB   = (bf16_t*)wsalloc((size_t)ROWS * DIM * 2);
  bf16_t* VT   = (bf16_t*)wsalloc((size_t)BATCH * DIM * SEQ * 2);
  bf16_t* OB   = (bf16_t*)wsalloc((size_t)ROWS * DIM * 2);
  bf16_t* CAT  = (bf16_t*)wsalloc((size_t)ROWS * CAT_DIM * 2);
  // PRE (f32, ROWS*DIM*4 = 18,874,368 B) aliases QB+KB (2 x 9,437,184 B,
  // contiguous, dead after the attention kernel of the same block).
  float* PRE = (float*)QB;

  // Weight conversion (once per call; deterministic)
  cvt_bf16_kernel<<<1024, 256, 0, stream>>>(Wq, wq_b, (int)WMAT);
  cvt_bf16_kernel<<<1024, 256, 0, stream>>>(Wk, wk_b, (int)WMAT);
  cvt_bf16_kernel<<<1024, 256, 0, stream>>>(Wv, wv_b, (int)WMAT);
  cvt_bf16_kernel<<<1024, 256, 0, stream>>>(Wp, wp_b, (int)WMAT);
  cvt_bf16_kernel<<<1024, 256, 0, stream>>>(Wf, wf_b, DIM * CAT_DIM);

  // pair order: (img,aolp),(img,dolp),(aolp,img),(aolp,dolp),(dolp,img),(dolp,aolp)
  const float* xsrc[NBLK] = {image, image, aolp, aolp, dolp, dolp};
  const float* ysrc[NBLK] = {aolp, dolp, image, dolp, image, aolp};

  const int GEMM_BLOCKS = (ROWS / 256) * (DIM / 64);    // 32*9 = 288

  for (int c = 0; c < NBLK; ++c) {
    const size_t po = (size_t)c * DIM;
    const size_t wo = (size_t)c * DIM * DIM;

    ln_kernel<<<ROWS / 8, 256, 0, stream>>>(xsrc[c], DIM, lnx_g + po,
                                            lnx_b + po, XN, DIM, STATS, ROWS);
    ln_kernel<<<ROWS / 8, 256, 0, stream>>>(ysrc[c], DIM, lny_g + po,
                                            lny_b + po, YN, DIM, nullptr, ROWS);

    gemm_wmma_kernel<0><<<GEMM_BLOCKS, 256, 0, stream>>>(
        YN, DIM, wq_b + wo, DIM, nullptr, nullptr, QB,
        nullptr, nullptr, nullptr, nullptr, DIM, DIM / 64);
    gemm_wmma_kernel<0><<<GEMM_BLOCKS, 256, 0, stream>>>(
        XN, DIM, wk_b + wo, DIM, nullptr, nullptr, KB,
        nullptr, nullptr, nullptr, nullptr, DIM, DIM / 64);
    gemm_wmma_kernel<1><<<GEMM_BLOCKS, 256, 0, stream>>>(
        XN, DIM, wv_b + wo, DIM, nullptr, nullptr, VT,
        nullptr, nullptr, nullptr, nullptr, DIM, DIM / 64);

    attn_kernel<<<dim3(SEQ / 16, BATCH), 256, 0, stream>>>(QB, KB, VT, OB);

    gemm_wmma_kernel<2><<<GEMM_BLOCKS, 256, 0, stream>>>(
        OB, DIM, wp_b + wo, DIM, bp + po, PRE, nullptr,
        xsrc[c], STATS, lnx_g + po, lnx_b + po, DIM, DIM / 64);

    ln_kernel<<<ROWS / 8, 256, 0, stream>>>(PRE, DIM, lnz_g + po, lnz_b + po,
                                            CAT + po, CAT_DIM, nullptr, ROWS);
  }

  // fused = gelu(concat @ Wf^T + bf)
  gemm_wmma_kernel<3><<<GEMM_BLOCKS, 256, 0, stream>>>(
      CAT, CAT_DIM, wf_b, CAT_DIM, bfv, (float*)d_out, nullptr,
      nullptr, nullptr, nullptr, nullptr, CAT_DIM, DIM / 64);
}
